// AttentionLayer_43361989820685
// MI455X (gfx1250) — matrix operand order repaired and checked by reading, not compiled, not run on hardware
//
#include <hip/hip_runtime.h>
#include <float.h>

// Problem constants (match reference)
#define NNODES 50000
#define NEDGES 400000
#define DIM    64
#define NHEADS 8
#define HDIM   512    // H*D
#define HIDDEN 1024
#define NEG_SLOPE 0.2f
#define BN_EPS 1e-5f

typedef __attribute__((ext_vector_type(16))) __bf16 v16bf;
typedef __attribute__((ext_vector_type(8)))  float  v8f;

// ---------- helpers ----------
__device__ __forceinline__ void atomicMaxFloat(float* addr, float val) {
  // standard sign-split trick; addr must be initialized to -FLT_MAX
  if (__float_as_int(val) >= 0)
    atomicMax((int*)addr, __float_as_int(val));
  else
    atomicMin((unsigned int*)addr, (unsigned int)__float_as_uint(val));
}

// ---------- generic fill ----------
__global__ void fill_f32(float* __restrict__ p, float v, long long n) {
  long long t = (long long)blockIdx.x * blockDim.x + threadIdx.x;
  if (t < n) p[t] = v;
}

// ---------- weight pre-convert: W[Nc,K] f32 -> WT[K,Nc] bf16 (k-major) ----------
__global__ void convert_wT_kernel(const float* __restrict__ W, __bf16* __restrict__ WT,
                                  int Nc, int K) {
  long long t = (long long)blockIdx.x * blockDim.x + threadIdx.x;
  if (t >= (long long)Nc * K) return;
  int n = (int)(t / K), k = (int)(t % K);          // read coalesced along k
  WT[(size_t)k * Nc + n] = (__bf16)W[t];
}

// ---------- WMMA bf16 GEMM: Out[M,Nc] = act(affine(A)[M,K] @ WT[K,Nc] + bias) ----------
// WT is bf16, k-major (pre-transposed). Each wave: 16 rows x 64 cols (4 accumulators).
// Block = 256 thr = 8 waves = 128 rows. Weight tiles (32K x 64N bf16 = 4KB) are moved
// global->LDS with async b128 copies (ASYNCcnt), double-buffered against the WMMAs.
// A fragment layout per ISA (16-bit A 16x32): v<4: K = 2v + hi*8 ; v>=4: K = 16+2(v-4)+hi*8.
// B fragment: N = lane&15, VGPR v packs the same K pair as A. C/D: col = lane&15, row = vgpr + hi*8.
#define LDS_WSTRIDE 72   // bf16 elems per k-row (144B: keeps 16B alignment for B128 writes)

__global__ __launch_bounds__(256)
void gemm_wmma_bf16(const float* __restrict__ A, const __bf16* __restrict__ WT,
                    const float* __restrict__ ascale, const float* __restrict__ ashift,
                    const float* __restrict__ bias, float* __restrict__ Out,
                    int M, int K, int Ncols, int relu) {
  __shared__ __bf16 ldsW[2][32 * LDS_WSTRIDE];
  const int tid  = threadIdx.x;
  const int lane = tid & 31;
  const int wave = tid >> 5;
  const int n0   = blockIdx.x * 64;
  const int m0   = (blockIdx.y * 8 + wave) * 16;
  const bool active = (m0 < M);
  const int hi   = lane >> 4;
  const int mrow = m0 + (lane & 15);
  const int kk   = tid >> 3;        // 0..31: k-row this thread stages
  const int seg  = tid & 7;         // 0..7 : 8-elem segment within the 64-col row

  v8f acc[4];
#pragma unroll
  for (int nt = 0; nt < 4; ++nt)
#pragma unroll
    for (int r = 0; r < 8; ++r) acc[nt][r] = 0.0f;

  // async-stage weight tile for k0 into buffer b (each thread copies 16 bytes)
  auto stage = [&](int k0, int b) {
    const __bf16* gp = WT + (size_t)(k0 + kk) * Ncols + n0 + seg * 8;
    unsigned lo = (unsigned)(uintptr_t)&ldsW[b][kk * LDS_WSTRIDE + seg * 8];
    asm volatile("global_load_async_to_lds_b128 %0, %1, off"
                 :: "v"(lo), "v"(gp) : "memory");
  };

  int buf = 0;
  stage(0, 0);
  for (int k0 = 0; k0 < K; k0 += 32) {
    const bool more = (k0 + 32) < K;
    if (more) {
      stage(k0 + 32, buf ^ 1);                       // prefetch next tile
      asm volatile("s_wait_asynccnt 0x1" ::: "memory");  // this tile's copy landed
    } else {
      asm volatile("s_wait_asynccnt 0x0" ::: "memory");
    }
    __syncthreads();                                 // all threads' segments visible

    if (active) {
      if (more) __builtin_prefetch(A + (size_t)mrow * K + k0 + 32, 0, 0);
      // A fragment (f32 -> bf16 native convert; optional fused BN affine)
      v16bf a;
#pragma unroll
      for (int v = 0; v < 8; ++v) {
        int koff = (v < 4) ? (2 * v + hi * 8) : (16 + 2 * (v - 4) + hi * 8);
        const float* ap = A + (size_t)mrow * K + k0 + koff;
        float a0 = ap[0], a1 = ap[1];
        if (ascale) {
          a0 = a0 * ascale[k0 + koff]     + ashift[k0 + koff];
          a1 = a1 * ascale[k0 + koff + 1] + ashift[k0 + koff + 1];
        }
        a[2 * v] = (__bf16)a0; a[2 * v + 1] = (__bf16)a1;
      }
#pragma unroll
      for (int nt = 0; nt < 4; ++nt) {
        v16bf b;
        const __bf16* wp = &ldsW[buf][nt * 16 + (lane & 15)];
#pragma unroll
        for (int v = 0; v < 8; ++v) {
          int kb = (v < 4) ? (2 * v + hi * 8) : (16 + 2 * (v - 4) + hi * 8);
          b[2 * v] = wp[kb * LDS_WSTRIDE]; b[2 * v + 1] = wp[(kb + 1) * LDS_WSTRIDE];
        }
        acc[nt] = __builtin_amdgcn_wmma_f32_16x16x32_bf16(
            false, a, false, b, (short)0, acc[nt], false, false);
      }
    }
    __syncthreads();                                 // done reading buf before reuse
    buf ^= 1;
  }

  if (active) {
#pragma unroll
    for (int nt = 0; nt < 4; ++nt) {
      int col = n0 + nt * 16 + (lane & 15);
      float bv = bias ? bias[col] : 0.0f;
#pragma unroll
      for (int r = 0; r < 8; ++r) {
        int row = m0 + r + hi * 8;
        float o = acc[nt][r] + bv;
        if (relu) o = fmaxf(o, 0.0f);
        Out[(size_t)row * Ncols + col] = o;
      }
    }
  }
}

// ---------- GAT attention scalars ----------
__global__ void eler_kernel(const float* __restrict__ feat,
                            const float* __restrict__ attn_l, const float* __restrict__ attn_r,
                            float* __restrict__ el, float* __restrict__ er, int NH) {
  int t = blockIdx.x * blockDim.x + threadIdx.x;
  if (t >= NH) return;
  int n = t >> 3, h = t & 7;
  const float* f  = feat + (size_t)n * HDIM + h * DIM;
  const float* al = attn_l + h * DIM;
  const float* ar = attn_r + h * DIM;
  float sl = 0.f, sr = 0.f;
#pragma unroll 8
  for (int d = 0; d < DIM; ++d) { float fv = f[d]; sl += fv * al[d]; sr += fv * ar[d]; }
  el[t] = sl; er[t] = sr;
}

__global__ void edge1_kernel(const float* __restrict__ el, const float* __restrict__ er,
                             const int* __restrict__ src, const int* __restrict__ dst,
                             float* __restrict__ ebuf, float* __restrict__ emax, int EH) {
  int t = blockIdx.x * blockDim.x + threadIdx.x;
  if (t >= EH) return;
  int e = t >> 3, h = t & 7;
  int s = src[e], d = dst[e];
  float v = el[s * NHEADS + h] + er[d * NHEADS + h];
  v = (v > 0.f) ? v : NEG_SLOPE * v;                     // leaky_relu
  ebuf[t] = v;
  atomicMaxFloat(&emax[d * NHEADS + h], v);
}

__global__ void edge2_kernel(const float* __restrict__ emax, const int* __restrict__ dst,
                             float* __restrict__ ebuf, float* __restrict__ denom, int EH) {
  int t = blockIdx.x * blockDim.x + threadIdx.x;
  if (t >= EH) return;
  int e = t >> 3, h = t & 7;
  int d = dst[e];
  float m = emax[d * NHEADS + h];
  if (m == -FLT_MAX) m = 0.f;                            // matches isfinite fixup
  float z = __expf(ebuf[t] - m);
  ebuf[t] = z;
  atomicAdd(&denom[d * NHEADS + h], z);
}

__global__ void edge3_kernel(const float* __restrict__ ebuf, const float* __restrict__ denom,
                             const float* __restrict__ feat,
                             const int* __restrict__ src, const int* __restrict__ dst,
                             float* __restrict__ rst, long long total) {
  long long t = (long long)blockIdx.x * blockDim.x + threadIdx.x;
  if (t >= total) return;
  int e = (int)(t >> 9);            // /512
  int col = (int)(t & 511);
  int h = col >> 6;                 // /64
  int s = src[e], d = dst[e];
  float alpha = ebuf[(size_t)e * NHEADS + h] / denom[(size_t)d * NHEADS + h];
  atomicAdd(&rst[(size_t)d * HDIM + col], alpha * feat[(size_t)s * HDIM + col]);
}

// ---------- BatchNorm (training mode, batch stats over rows) ----------
__global__ void bn_stats_kernel(const float* __restrict__ Hm, float* __restrict__ colsum,
                                float* __restrict__ colsq, int Nrows, int C) {
  int c = threadIdx.x;              // blockDim.x == C
  float s = 0.f, q = 0.f;
  for (int r = blockIdx.x; r < Nrows; r += gridDim.x) {
    float v = Hm[(size_t)r * C + c];
    s += v; q += v * v;
  }
  atomicAdd(&colsum[c], s);
  atomicAdd(&colsq[c], q);
}

__global__ void bn_final_kernel(const float* __restrict__ colsum, const float* __restrict__ colsq,
                                const float* __restrict__ gamma, const float* __restrict__ beta,
                                float* __restrict__ scale, float* __restrict__ shift,
                                int Nrows, int C) {
  int c = blockIdx.x * blockDim.x + threadIdx.x;
  if (c >= C) return;
  float inv = 1.0f / (float)Nrows;
  float mu  = colsum[c] * inv;
  float var = colsq[c] * inv - mu * mu;
  float sc  = gamma[c] * rsqrtf(var + BN_EPS);
  scale[c] = sc;
  shift[c] = beta[c] - mu * sc;
}

__global__ void bn_apply_kernel(float* __restrict__ Hm, const float* __restrict__ scale,
                                const float* __restrict__ shift, long long total, int C) {
  long long t = (long long)blockIdx.x * blockDim.x + threadIdx.x;
  if (t >= total) return;
  int c = (int)(t % C);
  Hm[t] = Hm[t] * scale[c] + shift[c];
}

// ---------- launch ----------
static inline unsigned blocksFor(long long n, int bs) { return (unsigned)((n + bs - 1) / bs); }

extern "C" void kernel_launch(void* const* d_in, const int* in_sizes, int n_in,
                              void* d_out, int out_size, void* d_ws, size_t ws_size,
                              hipStream_t stream) {
  const float* x        = (const float*)d_in[0];
  const int*   src      = (const int*)  d_in[1];
  const int*   dst      = (const int*)  d_in[2];
  const float* fc_w     = (const float*)d_in[3];
  const float* attn_l   = (const float*)d_in[4];
  const float* attn_r   = (const float*)d_in[5];
  // d_in[6] = gat_bias: a constant per-column shift; BN1 subtracts the batch
  // mean, so it cancels exactly -> elided.
  const float* bn1_gamma = (const float*)d_in[7];
  const float* bn1_beta  = (const float*)d_in[8];
  const float* w1        = (const float*)d_in[9];
  const float* b1        = (const float*)d_in[10];
  const float* w2        = (const float*)d_in[11];
  const float* b2        = (const float*)d_in[12];
  const float* bn2_gamma = (const float*)d_in[13];
  const float* bn2_beta  = (const float*)d_in[14];
  float* out = (float*)d_out;

  // workspace carve-up (floats, all chunk sizes multiples of 8 -> 16B alignment holds)
  float* ws = (float*)d_ws;
  size_t off = 0;
  float* feat  = ws + off; off += (size_t)NNODES * HDIM;   // 25.6M
  float* rst   = ws + off; off += (size_t)NNODES * HDIM;   // 25.6M
  float* h1    = ws + off; off += (size_t)NNODES * HIDDEN; // 51.2M
  float* el    = ws + off; off += (size_t)NNODES * NHEADS;
  float* er    = ws + off; off += (size_t)NNODES * NHEADS;
  float* ebuf  = ws + off; off += (size_t)NEDGES * NHEADS; // 3.2M
  float* emax  = ws + off; off += (size_t)NNODES * NHEADS;
  float* denom = ws + off; off += (size_t)NNODES * NHEADS;
  float* colsum = ws + off; off += 1024;
  float* colsq  = ws + off; off += 1024;
  float* scale  = ws + off; off += 1024;
  float* shift  = ws + off; off += 1024;
  // bf16 transposed weights
  __bf16* fcT = (__bf16*)(ws + off);            // [64 ,512 ] = 32768
  __bf16* w1T = fcT + (size_t)DIM * HDIM;       // [512,1024] = 524288
  __bf16* w2T = w1T + (size_t)HDIM * HIDDEN;    // [1024,64 ] = 65536
  (void)ws_size; (void)in_sizes; (void)n_in; (void)out_size;

  const int mtilesY = (NNODES / 16 + 7) / 8;   // 391 (50000 = 16*3125)

  // weight conversions (tiny; L2-resident afterwards)
  convert_wT_kernel<<<blocksFor((long long)HDIM * DIM, 256), 256, 0, stream>>>(
      fc_w, fcT, HDIM, DIM);
  convert_wT_kernel<<<blocksFor((long long)HIDDEN * HDIM, 256), 256, 0, stream>>>(
      w1, w1T, HIDDEN, HDIM);
  convert_wT_kernel<<<blocksFor((long long)DIM * HIDDEN, 256), 256, 0, stream>>>(
      w2, w2T, DIM, HIDDEN);

  // --- GATConv ---
  // feat = x @ fc_w^T  [50000,64]x[512,64]^T
  gemm_wmma_bf16<<<dim3(HDIM / 64, mtilesY), 256, 0, stream>>>(
      x, fcT, nullptr, nullptr, nullptr, feat, NNODES, DIM, HDIM, 0);

  eler_kernel<<<blocksFor((long long)NNODES * NHEADS, 256), 256, 0, stream>>>(
      feat, attn_l, attn_r, el, er, NNODES * NHEADS);

  fill_f32<<<blocksFor((long long)NNODES * NHEADS, 256), 256, 0, stream>>>(
      emax, -FLT_MAX, (long long)NNODES * NHEADS);
  fill_f32<<<blocksFor((long long)NNODES * NHEADS, 256), 256, 0, stream>>>(
      denom, 0.0f, (long long)NNODES * NHEADS);
  fill_f32<<<blocksFor((long long)NNODES * HDIM, 256), 256, 0, stream>>>(
      rst, 0.0f, (long long)NNODES * HDIM);

  edge1_kernel<<<blocksFor((long long)NEDGES * NHEADS, 256), 256, 0, stream>>>(
      el, er, src, dst, ebuf, emax, NEDGES * NHEADS);
  edge2_kernel<<<blocksFor((long long)NEDGES * NHEADS, 256), 256, 0, stream>>>(
      emax, dst, ebuf, denom, NEDGES * NHEADS);
  edge3_kernel<<<blocksFor((long long)NEDGES * HDIM, 256), 256, 0, stream>>>(
      ebuf, denom, feat, src, dst, rst, (long long)NEDGES * HDIM);

  // --- BN1 (affine folded into next GEMM's A-load) ---
  fill_f32<<<blocksFor(2048, 256), 256, 0, stream>>>(colsum, 0.0f, 2048); // colsum+colsq
  bn_stats_kernel<<<512, HDIM, 0, stream>>>(rst, colsum, colsq, NNODES, HDIM);
  bn_final_kernel<<<1, HDIM, 0, stream>>>(colsum, colsq, bn1_gamma, bn1_beta,
                                          scale, shift, NNODES, HDIM);

  // h1 = relu(BN1(rst) @ w1^T + b1)
  gemm_wmma_bf16<<<dim3(HIDDEN / 64, mtilesY), 256, 0, stream>>>(
      rst, w1T, scale, shift, b1, h1, NNODES, HDIM, HIDDEN, 1);

  // out = h1 @ w2^T + b2
  gemm_wmma_bf16<<<dim3(DIM / 64, mtilesY), 256, 0, stream>>>(
      h1, w2T, nullptr, nullptr, b2, out, NNODES, HIDDEN, DIM, 0);

  // --- BN2 applied in place on out ---
  fill_f32<<<blocksFor(2048, 256), 256, 0, stream>>>(colsum, 0.0f, 2048);
  bn_stats_kernel<<<512, DIM, 0, stream>>>(out, colsum, colsq, NNODES, DIM);
  bn_final_kernel<<<1, DIM, 0, stream>>>(colsum, colsq, bn2_gamma, bn2_beta,
                                         scale, shift, NNODES, DIM);
  bn_apply_kernel<<<blocksFor((long long)NNODES * DIM, 256), 256, 0, stream>>>(
      out, scale, shift, (long long)NNODES * DIM, DIM);
}
